// SimplifiedGraphTransformerLayer_64398739636996
// MI455X (gfx1250) — compile-verified, matrix operand order checked by reading, not run on hardware
//
#include <hip/hip_runtime.h>
#include <hip/hip_bf16.h>

// ---------------------------------------------------------------------------
// Types for CDNA5 WMMA (wave32): V_WMMA_F32_16X16X32_BF16
// ---------------------------------------------------------------------------
typedef __attribute__((ext_vector_type(16))) __bf16 v16bf;
typedef __attribute__((ext_vector_type(8)))  __bf16 v8bf;
typedef __attribute__((ext_vector_type(8)))  float  v8f;
typedef __attribute__((ext_vector_type(4)))  float  v4f;

static constexpr int Bq = 4, Nn = 2048, Dd = 512, Hh = 8, HDd = 64;

__device__ __forceinline__ v8f wmma_bf16(v16bf a, v16bf b, v8f c) {
    // D = A(16x32 bf16) * B(32x16 bf16) + C(16x16 f32)
    return __builtin_amdgcn_wmma_f32_16x16x32_bf16(
        /*neg_a=*/false, a, /*neg_b=*/false, b,
        /*c_mod=*/(short)0, c, /*reuse_a=*/false, /*reuse_b=*/false);
}

// Load one 16-bit A/B fragment half-pair from row-major storage.
// `base` points at element (row, k0 + hf*8); second chunk is +16 elements
// (= k0 + 16 + hf*8), matching the ISA 16-bit A-matrix 16x32 layout.
__device__ __forceinline__ v16bf load_bf16_frag(const __bf16* base) {
    union { v16bf v; v8bf h[2]; } u;
    u.h[0] = *(const v8bf*)(base);
    u.h[1] = *(const v8bf*)(base + 16);
    return u.v;
}

// ---------------------------------------------------------------------------
// Kernel 1: convert f32 weights -> bf16 transposed (WT[n][k] = W[k][n]).
// WT3 holds WqT|WkT|WvT concatenated; WoT separate.
// ---------------------------------------------------------------------------
__global__ __launch_bounds__(256) void convert_weights_kernel(
    const float* __restrict__ Wq, const float* __restrict__ Wk,
    const float* __restrict__ Wv, const float* __restrict__ Wo,
    __bf16* __restrict__ WT3, __bf16* __restrict__ WoT)
{
    int i = blockIdx.x * blockDim.x + threadIdx.x;      // i = k*512 + n
    if (i >= Dd * Dd) return;
    int k = i >> 9, n = i & (Dd - 1);
    size_t t = (size_t)n * Dd + k;
    WT3[t]                 = (__bf16)Wq[i];
    WT3[(size_t)Dd*Dd + t] = (__bf16)Wk[i];
    WT3[(size_t)2*Dd*Dd+t] = (__bf16)Wv[i];
    WoT[t]                 = (__bf16)Wo[i];
}

// ---------------------------------------------------------------------------
// Kernel 2: QKV projection.  C(8192x512) = X(8192x512,f32->bf16) @ W.
// gridDim.x = 512 row-tiles * 3 matrices; 8 waves/block, each wave a 16x64 tile.
// Q,K written bf16 [B*N][512]; V written transposed per head [B][H][HD][N].
// ---------------------------------------------------------------------------
__global__ __launch_bounds__(256) void qkv_proj_kernel(
    const float* __restrict__ x, const __bf16* __restrict__ WT3,
    const float* __restrict__ bq, const float* __restrict__ bk,
    const float* __restrict__ bv,
    __bf16* __restrict__ qb, __bf16* __restrict__ kb, __bf16* __restrict__ vT)
{
    const int lane = threadIdx.x & 31, wv = threadIdx.x >> 5;
    const int l16 = lane & 15, hf = lane >> 4;
    const int mt   = blockIdx.x & 511;      // row tile (16 rows)
    const int wsel = blockIdx.x >> 9;       // 0=Q 1=K 2=V
    const int m0 = mt * 16;
    const int cg = wv * 64;                 // 64-col group inside the matrix
    const __bf16* Wt = WT3 + (size_t)wsel * Dd * Dd;
    const float* bias = (wsel == 0) ? bq : (wsel == 1 ? bk : bv);

    v8f acc[4] = {};
    const float* xr = x + (size_t)(m0 + l16) * Dd + hf * 8;

    for (int k0 = 0; k0 < Dd; k0 += 32) {
        // A fragment: convert 16 f32 -> bf16 on the fly
        v4f x0 = *(const v4f*)(xr + k0);
        v4f x1 = *(const v4f*)(xr + k0 + 4);
        v4f x2 = *(const v4f*)(xr + k0 + 16);
        v4f x3 = *(const v4f*)(xr + k0 + 20);
        v16bf a;
#pragma unroll
        for (int e = 0; e < 4; ++e) {
            a[e]      = (__bf16)x0[e];
            a[4 + e]  = (__bf16)x1[e];
            a[8 + e]  = (__bf16)x2[e];
            a[12 + e] = (__bf16)x3[e];
        }
#pragma unroll
        for (int c = 0; c < 4; ++c) {
            v16bf bfr = load_bf16_frag(Wt + (size_t)(cg + c*16 + l16) * Dd + k0 + hf * 8);
            acc[c] = wmma_bf16(a, bfr, acc[c]);
        }
    }

#pragma unroll
    for (int c = 0; c < 4; ++c) {
        const int col = cg + c * 16 + l16;
        const float bi = bias[col];
#pragma unroll
        for (int r = 0; r < 8; ++r) {
            const int row = m0 + r + 8 * hf;          // global (b*N+n)
            const float val = acc[c][r] + bi;
            if (wsel == 0) {
                qb[(size_t)row * Dd + col] = (__bf16)val;
            } else if (wsel == 1) {
                kb[(size_t)row * Dd + col] = (__bf16)val;
            } else {
                const int bidx = row >> 11, n = row & (Nn - 1);
                const int h = col >> 6, hd = col & (HDd - 1);
                vT[(((size_t)bidx * Hh + h) * HDd + hd) * Nn + n] = (__bf16)val;
            }
        }
    }
}

// ---------------------------------------------------------------------------
// Kernel 3: masked flash attention.
// grid = B*H*(N/64); 4 waves/block, each wave owns a 16-row q tile.
// Online softmax; probs transposed acc-layout -> A-layout through LDS.
// ---------------------------------------------------------------------------
__global__ __launch_bounds__(128) void attn_kernel(
    const __bf16* __restrict__ qb, const __bf16* __restrict__ kb,
    const __bf16* __restrict__ vT, const float* __restrict__ adj,
    __bf16* __restrict__ ob)
{
    __shared__ __align__(16) __bf16 pstage[4][16][32];   // 4 KB, 1 KB per wave

    const int lane = threadIdx.x & 31, wv = threadIdx.x >> 5;
    const int l16 = lane & 15, hf = lane >> 4;
    int idx = blockIdx.x;
    const int qt = idx & 31; idx >>= 5;
    const int h  = idx & 7;  idx >>= 3;
    const int b  = idx;
    const int q0 = qt * 64 + wv * 16;

    // Q fragments (HD=64 -> 2 A fragments over K)
    const __bf16* qr = qb + ((size_t)(b * Nn + q0 + l16)) * Dd + h * HDd + hf * 8;
    const v16bf qf0 = load_bf16_frag(qr);
    const v16bf qf1 = load_bf16_frag(qr + 32);
    const __bf16* kbase = kb + ((size_t)b * Nn) * Dd + h * HDd + hf * 8;
    const __bf16* vbase = vT + (((size_t)b * Hh + h) * HDd) * Nn;

    v8f o[4] = {};
    float mrow[8], lrow[8];
#pragma unroll
    for (int r = 0; r < 8; ++r) { mrow[r] = -1e30f; lrow[r] = 0.0f; }

    for (int kj = 0; kj < Nn; kj += 32) {
        // scores: two 16x16 f32 fragments (cols kj..+15, kj+16..+31)
        v8f s0 = {}, s1 = {};
        {
            const __bf16* kr0 = kbase + (size_t)(kj + l16) * Dd;
            const __bf16* kr1 = kr0 + (size_t)16 * Dd;
            s0 = wmma_bf16(qf0, load_bf16_frag(kr0),      s0);
            s0 = wmma_bf16(qf1, load_bf16_frag(kr0 + 32), s0);
            s1 = wmma_bf16(qf0, load_bf16_frag(kr1),      s1);
            s1 = wmma_bf16(qf1, load_bf16_frag(kr1 + 32), s1);
        }

        // mask + online softmax (per-row stats replicated across 16-lane group)
#pragma unroll
        for (int r = 0; r < 8; ++r) {
            const int row = q0 + r + 8 * hf;
            const float a0 = adj[(size_t)row * Nn + kj + l16];
            const float a1 = adj[(size_t)row * Nn + kj + 16 + l16];
            float v0 = s0[r] * 0.125f + (1.0f - a0) * (-1e9f);  // 1/sqrt(64)
            float v1 = s1[r] * 0.125f + (1.0f - a1) * (-1e9f);
            float mx = fmaxf(v0, v1);
            mx = fmaxf(mx, __shfl_xor(mx, 1));
            mx = fmaxf(mx, __shfl_xor(mx, 2));
            mx = fmaxf(mx, __shfl_xor(mx, 4));
            mx = fmaxf(mx, __shfl_xor(mx, 8));
            const float mn = fmaxf(mrow[r], mx);
            const float c0 = __expf(mrow[r] - mn);
            const float p0 = __expf(v0 - mn);
            const float p1 = __expf(v1 - mn);
            float rs = p0 + p1;
            rs += __shfl_xor(rs, 1);
            rs += __shfl_xor(rs, 2);
            rs += __shfl_xor(rs, 4);
            rs += __shfl_xor(rs, 8);
            lrow[r] = lrow[r] * c0 + rs;
            mrow[r] = mn;
#pragma unroll
            for (int c = 0; c < 4; ++c) o[c][r] *= c0;
            // stage probs (accumulator layout) to LDS for fragment transpose
            pstage[wv][r + 8 * hf][l16]      = (__bf16)p0;
            pstage[wv][r + 8 * hf][l16 + 16] = (__bf16)p1;
        }
        __syncthreads();

        // reload probs as A fragment; accumulate O += P @ V
        const v16bf pA = load_bf16_frag(&pstage[wv][l16][hf * 8]);
#pragma unroll
        for (int c = 0; c < 4; ++c) {
            v16bf vf = load_bf16_frag(vbase + (size_t)(c * 16 + l16) * Nn + kj + hf * 8);
            o[c] = wmma_bf16(pA, vf, o[c]);
        }
        __syncthreads();
    }

    // normalize and write O (bf16, [B*N][512])
#pragma unroll
    for (int c = 0; c < 4; ++c) {
#pragma unroll
        for (int r = 0; r < 8; ++r) {
            const float val = o[c][r] / lrow[r];
            ob[((size_t)(b * Nn + q0 + r + 8 * hf)) * Dd + h * HDd + c * 16 + l16] =
                (__bf16)val;
        }
    }
}

// ---------------------------------------------------------------------------
// Kernel 4: output projection.  out = O @ Wo + bo  (f32 result).
// 512 blocks, 8 waves/block, each wave a 16x64 tile.
// ---------------------------------------------------------------------------
__global__ __launch_bounds__(256) void out_proj_kernel(
    const __bf16* __restrict__ ob, const __bf16* __restrict__ WoT,
    const float* __restrict__ bo, float* __restrict__ out)
{
    const int lane = threadIdx.x & 31, wv = threadIdx.x >> 5;
    const int l16 = lane & 15, hf = lane >> 4;
    const int m0 = blockIdx.x * 16;
    const int cg = wv * 64;

    v8f acc[4] = {};
    const __bf16* ar = ob + (size_t)(m0 + l16) * Dd + hf * 8;
    for (int k0 = 0; k0 < Dd; k0 += 32) {
        v16bf a = load_bf16_frag(ar + k0);
#pragma unroll
        for (int c = 0; c < 4; ++c) {
            v16bf bfr = load_bf16_frag(WoT + (size_t)(cg + c*16 + l16) * Dd + k0 + hf * 8);
            acc[c] = wmma_bf16(a, bfr, acc[c]);
        }
    }
#pragma unroll
    for (int c = 0; c < 4; ++c) {
        const int col = cg + c * 16 + l16;
        const float bi = bo[col];
#pragma unroll
        for (int r = 0; r < 8; ++r)
            out[(size_t)(m0 + r + 8 * hf) * Dd + col] = acc[c][r] + bi;
    }
}

// ---------------------------------------------------------------------------
// Launch
// ---------------------------------------------------------------------------
extern "C" void kernel_launch(void* const* d_in, const int* in_sizes, int n_in,
                              void* d_out, int out_size, void* d_ws, size_t ws_size,
                              hipStream_t stream) {
    const float* x   = (const float*)d_in[0];
    const float* adj = (const float*)d_in[1];
    const float* Wq  = (const float*)d_in[2];
    const float* bq  = (const float*)d_in[3];
    const float* Wk  = (const float*)d_in[4];
    const float* bk  = (const float*)d_in[5];
    const float* Wv  = (const float*)d_in[6];
    const float* bv  = (const float*)d_in[7];
    const float* Wo  = (const float*)d_in[8];
    const float* bo  = (const float*)d_in[9];
    float* out = (float*)d_out;

    // workspace carve-up (bf16 elements)
    __bf16* WT3 = (__bf16*)d_ws;                       // 3 * 512*512
    __bf16* WoT = WT3 + (size_t)3 * Dd * Dd;           // 512*512
    __bf16* qb  = WoT + (size_t)Dd * Dd;               // 8192*512
    __bf16* kb  = qb  + (size_t)Bq * Nn * Dd;
    __bf16* vT  = kb  + (size_t)Bq * Nn * Dd;
    __bf16* ob  = vT  + (size_t)Bq * Nn * Dd;

    convert_weights_kernel<<<(Dd * Dd + 255) / 256, 256, 0, stream>>>(
        Wq, Wk, Wv, Wo, WT3, WoT);

    qkv_proj_kernel<<<512 * 3, 256, 0, stream>>>(
        x, WT3, bq, bk, bv, qb, kb, vT);

    attn_kernel<<<Bq * Hh * (Nn / 64), 128, 0, stream>>>(
        qb, kb, vT, adj, ob);

    out_proj_kernel<<<(Bq * Nn) / 16, 256, 0, stream>>>(
        ob, WoT, bo, out);
}
